// GCN_45586782880287
// MI455X (gfx1250) — compile-verified, matrix operand order checked by reading, not run on hardware
//
#include <hip/hip_runtime.h>
#include <hip/hip_bf16.h>
#include <math.h>

// ---------------------------------------------------------------------------
// GCN forward for MI455X (gfx1250), wave32 + WMMA.
//
// Pipeline (matches jax reference):
//   deg[d] = 1 + #edges into d ; dis = rsqrt(deg)
//   h  = x @ W1                          (WMMA f16->f32)
//   a1 = seg_sum(h[s]*dis[s]*dis[d]) + h*dis^2  ; relu(a1 + b1)
//   h2 = relu_out @ W2                   (WMMA)
//   a2 = same aggregation ; + b2
//   g  = segment_max(a2, batch) ; out = g @ Wlin + blin
//
// WMMA register-layout assumptions (CDNA5 ISA 7.12.2, wave32):
//   A (16x32 f16):  lane r=lane&15, hi=lane>>4 holds row r halves
//                   [hi*8 .. hi*8+7] and [16+hi*8 .. 16+hi*8+7]  -> two b128 loads
//   B (32x16 f16):  mirrored: lane holds column r, K pairs {hi*8+2j,(+1)} j<4,
//                   {16+hi*8+2j'} j>=4 -> pre-packed by GCN_packB
//   C/D (16x16 f32): elem i of v8f at row (i + 8*hi), col r
// ---------------------------------------------------------------------------

typedef __attribute__((ext_vector_type(16))) _Float16 v16h;
typedef __attribute__((ext_vector_type(8)))  _Float16 v8h;
typedef __attribute__((ext_vector_type(8)))  float    v8f;

// ---------------- elementwise / init kernels ----------------

__global__ void GCN_fill_f32(float* __restrict__ p, float v, int n) {
  int i = blockIdx.x * blockDim.x + threadIdx.x;
  if (i < n) p[i] = v;
}

__global__ void GCN_fill_u32(unsigned* __restrict__ p, unsigned v, int n) {
  int i = blockIdx.x * blockDim.x + threadIdx.x;
  if (i < n) p[i] = v;
}

__global__ void GCN_edge_deg(const int* __restrict__ dst, float* __restrict__ deg, int E) {
  int i = blockIdx.x * blockDim.x + threadIdx.x;
  if (i < E) atomicAdd(&deg[dst[i]], 1.0f);
}

__global__ void GCN_rsqrt(const float* __restrict__ deg, float* __restrict__ dis, int n) {
  int i = blockIdx.x * blockDim.x + threadIdx.x;
  if (i < n) dis[i] = rsqrtf(deg[i]);   // deg >= 1 always (self loop)
}

__global__ void GCN_cvt16(const float* __restrict__ x, _Float16* __restrict__ o, int n) {
  int i = blockIdx.x * blockDim.x + threadIdx.x;
  if (i < n) o[i] = (_Float16)x[i];
}

// acc[n][f] = h[n][f] * dis[n]^2   (fused self-loop contribution = aggregator init)
__global__ void GCN_self_init(const float* __restrict__ h, const float* __restrict__ dis,
                              float* __restrict__ acc, int n64) {
  int i = blockIdx.x * blockDim.x + threadIdx.x;
  if (i >= n64) return;
  float ds = dis[i >> 6];
  acc[i] = h[i] * ds * ds;
}

// out[n][f] = f16(relu(acc[n][f] + b[f]))
__global__ void GCN_bias_relu16(const float* __restrict__ acc, const float* __restrict__ b,
                                _Float16* __restrict__ o, int n64) {
  int i = blockIdx.x * blockDim.x + threadIdx.x;
  if (i >= n64) return;
  float v = acc[i] + b[i & 63];
  o[i] = (_Float16)fmaxf(v, 0.0f);
}

// ---------------- weight pre-pack into WMMA B-fragment layout ----------------
// Bp[(nt*2+c)*32 + lane][0..15]  (f16), one 256-thread block, thread = (nt,c,lane)
__global__ void GCN_packB(const float* __restrict__ W, _Float16* __restrict__ Bp) {
  int t    = threadIdx.x;            // 0..255
  int lane = t & 31;
  int c    = (t >> 5) & 1;           // K chunk (0: K 0..31, 1: K 32..63)
  int nt   = t >> 6;                 // N tile
  int r    = lane & 15;
  int hi   = lane >> 4;
  int col  = nt * 16 + r;
  _Float16* o = Bp + (size_t)t * 16;
#pragma unroll
  for (int j = 0; j < 8; ++j) {
    int k = c * 32 + ((j < 4) ? (hi * 8 + 2 * j) : (16 + hi * 8 + 2 * (j - 4)));
    o[2 * j]     = (_Float16)W[(size_t)k * 64 + col];
    o[2 * j + 1] = (_Float16)W[(size_t)(k + 1) * 64 + col];
  }
}

// ---------------- WMMA GEMM: C[nrows x 64] = A[nrows x 64] * W(packed) -------
__global__ void GCN_gemm64(const _Float16* __restrict__ A, const _Float16* __restrict__ Bp,
                           float* __restrict__ C, int nrows) {
  int wave = blockIdx.x * (blockDim.x >> 5) + (threadIdx.x >> 5);
  int row0 = wave << 4;
  if (row0 >= nrows) return;                 // uniform per-wave exit
  int lane = threadIdx.x & 31;
  int r    = lane & 15;
  int hi   = lane >> 4;

  const _Float16* arow = A + (size_t)(row0 + r) * 64;
  v8h a00 = *(const v8h*)(arow +  0 + hi * 8);   // K  0..15 span
  v8h a01 = *(const v8h*)(arow + 16 + hi * 8);   // K 16..31 span
  v8h a10 = *(const v8h*)(arow + 32 + hi * 8);   // K 32..47 span
  v8h a11 = *(const v8h*)(arow + 48 + hi * 8);   // K 48..63 span
  v16h a0 = __builtin_shufflevector(a00, a01, 0,1,2,3,4,5,6,7,8,9,10,11,12,13,14,15);
  v16h a1 = __builtin_shufflevector(a10, a11, 0,1,2,3,4,5,6,7,8,9,10,11,12,13,14,15);

#pragma unroll
  for (int nt = 0; nt < 4; ++nt) {
    v16h b0 = *(const v16h*)(Bp + ((size_t)(nt * 2 + 0) * 32 + lane) * 16);
    v16h b1 = *(const v16h*)(Bp + ((size_t)(nt * 2 + 1) * 32 + lane) * 16);
    v8f acc = {};
    acc = __builtin_amdgcn_wmma_f32_16x16x32_f16(false, a0, false, b0,
                                                 (short)0, acc, false, false);
    acc = __builtin_amdgcn_wmma_f32_16x16x32_f16(false, a1, false, b1,
                                                 (short)0, acc, false, false);
    float* crow = C + (size_t)row0 * 64 + nt * 16 + r;
#pragma unroll
    for (int i = 0; i < 8; ++i)
      crow[(size_t)(i + hi * 8) * 64] = acc[i];
  }
}

// ---------------- edge gather/scale/scatter-add: one wave per edge ----------
__global__ void GCN_scatter(const float* __restrict__ h, const int* __restrict__ src,
                            const int* __restrict__ dst, const float* __restrict__ dis,
                            float* __restrict__ acc, int nedges) {
  int e = blockIdx.x * (blockDim.x >> 5) + (threadIdx.x >> 5);
  if (e >= nedges) return;
  int lane = threadIdx.x & 31;
  int s = __builtin_amdgcn_readfirstlane(src[e]);
  int d = __builtin_amdgcn_readfirstlane(dst[e]);
  float nrm = dis[s] * dis[d];
  const float2 v = *(const float2*)(h + (size_t)s * 64 + lane * 2);
  float* o = acc + (size_t)d * 64 + lane * 2;
  atomicAdd(o,     v.x * nrm);
  atomicAdd(o + 1, v.y * nrm);
}

// ---------------- segment max (batch sorted, but done with atomics) ---------
__global__ void GCN_segmax(const float* __restrict__ acc, const float* __restrict__ b,
                           const int* __restrict__ batch, unsigned* __restrict__ g, int n64) {
  int i = blockIdx.x * blockDim.x + threadIdx.x;
  if (i >= n64) return;
  int f = i & 63;
  float v = acc[i] + b[f];
  unsigned* addr = &g[(size_t)batch[i >> 6] * 64 + f];
  // lock-free float max: ordered int compare for >=0, unsigned min for <0
  if (v >= 0.0f) atomicMax((int*)addr, __float_as_int(v));
  else           atomicMin(addr, __float_as_uint(v));
}

// ---------------- final tiny classifier: out[G x 2] = g @ Wlin + blin -------
__global__ void GCN_final(const float* __restrict__ g, const float* __restrict__ Wl,
                          const float* __restrict__ bl, float* __restrict__ out, int G) {
  int t = blockIdx.x * blockDim.x + threadIdx.x;
  if (t >= G * 2) return;
  int gi = t >> 1, c = t & 1;
  float s = bl[c];
  const float* grow = g + (size_t)gi * 64;
#pragma unroll
  for (int f = 0; f < 64; ++f) s += grow[f] * Wl[f * 2 + c];
  out[(size_t)gi * 2 + c] = s;
}

// ---------------------------------------------------------------------------

extern "C" void kernel_launch(void* const* d_in, const int* in_sizes, int n_in,
                              void* d_out, int out_size, void* d_ws, size_t ws_size,
                              hipStream_t stream) {
  const float* x    = (const float*)d_in[0];
  const int*   ei   = (const int*)  d_in[1];
  const int*   bat  = (const int*)  d_in[2];
  const float* W1   = (const float*)d_in[3];
  const float* b1   = (const float*)d_in[4];
  const float* W2   = (const float*)d_in[5];
  const float* b2   = (const float*)d_in[6];
  const float* Wlin = (const float*)d_in[7];
  const float* blin = (const float*)d_in[8];
  float* out = (float*)d_out;

  const int N   = in_sizes[0] / 64;   // 50000
  const int E   = in_sizes[1] / 2;    // 800000
  const int G   = out_size / 2;       // 64 graphs
  const int N64 = N * 64;
  const int* src = ei;
  const int* dst = ei + E;

  // ----- carve workspace (256B aligned) -----
  char* ws = (char*)d_ws;
  size_t off = 0;
  auto carve = [&](size_t bytes) -> void* {
    void* p = ws + off;
    off += (bytes + 255) & ~(size_t)255;
    return p;
  };
  float*     deg = (float*)    carve((size_t)N * 4);
  float*     dis = (float*)    carve((size_t)N * 4);
  _Float16*  xh  = (_Float16*) carve((size_t)N64 * 2);   // f16 activations (layer in)
  _Float16*  hh  = (_Float16*) carve((size_t)N64 * 2);   // f16 activations (layer 2 in)
  float*     t   = (float*)    carve((size_t)N64 * 4);   // GEMM output (reused)
  float*     acc = (float*)    carve((size_t)N64 * 4);   // aggregation (reused)
  _Float16*  Bp1 = (_Float16*) carve(4096 * 2);
  _Float16*  Bp2 = (_Float16*) carve(4096 * 2);
  unsigned*  g   = (unsigned*) carve((size_t)G * 64 * 4);
  (void)ws_size; (void)n_in;

  const int TB = 256;
  dim3 blk(TB);
  int gN   = (N   + TB - 1) / TB;
  int gE   = (E   + TB - 1) / TB;
  int gN64 = (N64 + TB - 1) / TB;
  int gemmBlocks = (((N + 15) / 16) + 7) / 8;   // 8 waves/block, 16 rows/wave
  int edgeBlocks = (E + 7) / 8;                 // 8 edges/block (1 wave each)

  // degrees + symmetric norm
  GCN_fill_f32<<<gN, blk, 0, stream>>>(deg, 1.0f, N);
  GCN_edge_deg<<<gE, blk, 0, stream>>>(dst, deg, E);
  GCN_rsqrt<<<gN, blk, 0, stream>>>(deg, dis, N);

  // f16 conversions / weight packing
  GCN_cvt16<<<gN64, blk, 0, stream>>>(x, xh, N64);
  GCN_packB<<<1, blk, 0, stream>>>(W1, Bp1);
  GCN_packB<<<1, blk, 0, stream>>>(W2, Bp2);

  // ----- layer 1 -----
  GCN_gemm64<<<gemmBlocks, blk, 0, stream>>>(xh, Bp1, t, N);
  GCN_self_init<<<gN64, blk, 0, stream>>>(t, dis, acc, N64);
  GCN_scatter<<<edgeBlocks, blk, 0, stream>>>(t, src, dst, dis, acc, E);
  GCN_bias_relu16<<<gN64, blk, 0, stream>>>(acc, b1, hh, N64);

  // ----- layer 2 -----
  GCN_gemm64<<<gemmBlocks, blk, 0, stream>>>(hh, Bp2, t, N);
  GCN_self_init<<<gN64, blk, 0, stream>>>(t, dis, acc, N64);
  GCN_scatter<<<edgeBlocks, blk, 0, stream>>>(t, src, dst, dis, acc, E);

  // ----- pool + classify -----
  int gG = (G * 64 + TB - 1) / TB;
  GCN_fill_u32<<<gG, blk, 0, stream>>>(g, 0xFF800000u /* -inf */, G * 64);
  GCN_segmax<<<gN64, blk, 0, stream>>>(acc, b2, bat, g, N64);
  GCN_final<<<(G * 2 + TB - 1) / TB, blk, 0, stream>>>((const float*)g, Wlin, blin, out, G);
}
